// Decoder_24910810316801
// MI455X (gfx1250) — matrix operand order repaired and checked by reading, not compiled, not run on hardware
//
#include <hip/hip_runtime.h>
#include <math.h>

// Problem dims (match reference)
#define V_   32000
#define E_   512
#define H2_  1024
#define B_   32
#define S_   64
#define T_   32
#define X_   (H2_ + E_)   // 1536
#define G4_  (4 * H2_)    // 4096

typedef __attribute__((ext_vector_type(16))) __bf16 v16bf;
typedef __attribute__((ext_vector_type(8)))  float  v8f;

// ---------------- helpers ----------------

__device__ __forceinline__ unsigned short f32_to_bf16(float f) {
  unsigned int u = __float_as_uint(f);
  u += 0x7fffu + ((u >> 16) & 1u);          // round-to-nearest-even
  return (unsigned short)(u >> 16);
}

// A tile 16x32 (MxK), bf16, row-major source with leading dim lda.
// Lane<16: M=lane, K = k0..k0+15 contiguous; lane>=16: M=lane-16, K = k0+16..k0+31
__device__ __forceinline__ v16bf ld_a_16x32(const __bf16* A, int lda, int m0, int k0, int lane) {
  const int hi = lane >> 4;
  const int m  = m0 + (lane & 15);
  const uint4* p = reinterpret_cast<const uint4*>(A + (size_t)m * lda + k0 + hi * 16);
  union { v16bf v; uint4 q[2]; } u;
  u.q[0] = p[0];   // 8 bf16 at k0 (+16 for hi lanes)
  u.q[1] = p[1];   // 8 bf16 at k0+8 (+16 for hi lanes)
  return u.v;
}

// B tile 32x16 (KxN) from row-major weight W[N][K] (logical B[k][n] = W[n][k]).
// Lane<16: N=lane, K = k0..k0+15 contiguous; lane>=16: N=lane-16, K = k0+16..k0+31.
__device__ __forceinline__ v16bf ld_b_32x16(const __bf16* W, int ldk, int n0, int k0, int lane) {
  const int hi = lane >> 4;
  const int n  = n0 + (lane & 15);
  const uint4* p = reinterpret_cast<const uint4*>(W + (size_t)n * ldk + k0 + hi * 16);
  union { v16bf v; uint4 q[2]; } u;
  u.q[0] = p[0];
  u.q[1] = p[1];
  return u.v;
}

// One 16-row A stripe vs NB adjacent 16-col weight stripes; A loaded once per k-step.
template <int NB>
__device__ __forceinline__ void gemm_kloop(const __bf16* __restrict__ A, int lda, int m0,
                                           const __bf16* __restrict__ W, int ldk, int n0,
                                           int K, int lane, v8f (&acc)[NB]) {
  for (int k0 = 0; k0 < K; k0 += 32) {
    const v16bf a = ld_a_16x32(A, lda, m0, k0, lane);
#pragma unroll
    for (int j = 0; j < NB; ++j) {
      const v16bf b = ld_b_32x16(W, ldk, n0 + 16 * j, k0, lane);
      acc[j] = __builtin_amdgcn_wmma_f32_16x16x32_bf16(false, a, false, b, (short)0,
                                                       acc[j], false, false);
    }
  }
}

// ---------------- one-time kernels ----------------

__global__ void k_cvt_bf16(const float* __restrict__ src, unsigned short* __restrict__ dst, int n) {
  int i = blockIdx.x * blockDim.x + threadIdx.x;
  const int stride = gridDim.x * blockDim.x;
  for (; i < n; i += stride) dst[i] = f32_to_bf16(src[i]);
}

__global__ void k_init_state(const float* __restrict__ h0, const float* __restrict__ c0,
                             float* __restrict__ h, float* __restrict__ c,
                             unsigned short* __restrict__ hbf) {
  int i = blockIdx.x * blockDim.x + threadIdx.x;
  if (i < B_ * H2_) {
    float hv = h0[i];
    h[i] = hv;
    c[i] = c0[i];
    hbf[i] = f32_to_bf16(hv);
  }
}

// att_proj[m, e] = sum_d all_hidden[m, d] * W_att[e, d] + b_att[e];  m = b*S+s
__global__ void k_attproj_gemm(const __bf16* __restrict__ AH, const __bf16* __restrict__ Watt,
                               const float* __restrict__ b_att, float* __restrict__ attp) {
  const int lane = threadIdx.x & 31;
  const int n0 = blockIdx.x * 64;
  const int m0 = blockIdx.y * 16;
  v8f acc[4] = {};
  gemm_kloop<4>(AH, H2_, m0, Watt, H2_, n0, H2_, lane, acc);
  const int rb = m0 + ((lane >> 4) * 8);
#pragma unroll
  for (int j = 0; j < 4; ++j) {
    const int col = n0 + 16 * j + (lane & 15);
    const float bias = b_att[col];
#pragma unroll
    for (int r = 0; r < 8; ++r)
      attp[(size_t)(rb + r) * H2_ + col] = acc[j][r] + bias;
  }
}

// ---------------- per-step kernels ----------------

// One block per batch element; scores->softmax->ctx; also writes teacher-forcing feed.
__global__ void k_attention(const float* __restrict__ attp, const float* __restrict__ all_hidden,
                            const float* __restrict__ h, const float* __restrict__ emb,
                            const int* __restrict__ tgt, unsigned short* __restrict__ Xb, int t) {
  const int b = blockIdx.x;
  const int tid = threadIdx.x;              // 256 threads
  __shared__ float sh_h[H2_];
  __shared__ float sh_sc[S_];
  __shared__ float sh_part[256];

  for (int e = tid; e < H2_; e += 256) sh_h[e] = h[b * H2_ + e];
  __syncthreads();

  // scores[s] = att_proj[b,s,:] . h[b,:]   (4 threads per s)
  const int s = tid >> 2, q = tid & 3;
  {
    const float* row = attp + ((size_t)b * S_ + s) * H2_ + q * 256;
    float p = 0.f;
    for (int j = 0; j < 256; ++j) p += row[j] * sh_h[q * 256 + j];
    sh_part[tid] = p;
  }
  __syncthreads();
  if (q == 0)
    sh_sc[s] = sh_part[tid] + sh_part[tid + 1] + sh_part[tid + 2] + sh_part[tid + 3];
  __syncthreads();

  if (tid == 0) {
    float mx = -3.0e38f;
    for (int i = 0; i < S_; ++i) mx = fmaxf(mx, sh_sc[i]);
    float sum = 0.f;
    for (int i = 0; i < S_; ++i) { float ev = expf(sh_sc[i] - mx); sh_sc[i] = ev; sum += ev; }
    const float inv = 1.f / sum;
    for (int i = 0; i < S_; ++i) sh_sc[i] *= inv;
  }
  __syncthreads();

  // ctx[d] = sum_s p[s] * all_hidden[b,s,d] -> bf16 into x[:, 0:H2]
  for (int d = tid; d < H2_; d += 256) {
    const float* ah = all_hidden + (size_t)b * S_ * H2_ + d;
    float acc = 0.f;
    for (int si = 0; si < S_; ++si) acc += sh_sc[si] * ah[(size_t)si * H2_];
    Xb[(size_t)b * X_ + d] = f32_to_bf16(acc);
  }
  // feed = (t==0) ? 0 : embedding[target[b, t-1]] -> bf16 into x[:, H2:H2+E]
  for (int e = tid; e < E_; e += 256) {
    float f = 0.f;
    if (t > 0) {
      const int w = tgt[b * T_ + (t - 1)];
      f = emb[(size_t)w * E_ + e];
    }
    Xb[(size_t)b * X_ + H2_ + e] = f32_to_bf16(f);
  }
}

// gates[32, 4096] = x@W_ih^T + h@W_hh^T + b_ih + b_hh
__global__ void k_gates_gemm(const __bf16* __restrict__ Xb, const __bf16* __restrict__ Hb,
                             const __bf16* __restrict__ Wih, const __bf16* __restrict__ Whh,
                             const float* __restrict__ b_ih, const float* __restrict__ b_hh,
                             float* __restrict__ gates) {
  const int lane = threadIdx.x & 31;
  const int n0 = blockIdx.x * 64;
  const int m0 = blockIdx.y * 16;
  v8f acc[4] = {};
  gemm_kloop<4>(Xb, X_, m0, Wih, X_, n0, X_, lane, acc);    // x @ W_ih^T
  gemm_kloop<4>(Hb, H2_, m0, Whh, H2_, n0, H2_, lane, acc); // + h @ W_hh^T
  const int rb = m0 + ((lane >> 4) * 8);
#pragma unroll
  for (int j = 0; j < 4; ++j) {
    const int col = n0 + 16 * j + (lane & 15);
    const float bias = b_ih[col] + b_hh[col];
#pragma unroll
    for (int r = 0; r < 8; ++r)
      gates[(size_t)(rb + r) * G4_ + col] = acc[j][r] + bias;
  }
}

__global__ void k_lstm(const float* __restrict__ gates, float* __restrict__ h,
                       float* __restrict__ c, unsigned short* __restrict__ hbf) {
  const int b = blockIdx.x;
  const int tid = threadIdx.x;
  const float* gr = gates + (size_t)b * G4_;
  for (int g = tid; g < H2_; g += 256) {
    const float ig = gr[g], fg = gr[H2_ + g], gg = gr[2 * H2_ + g], og = gr[3 * H2_ + g];
    const float si = 1.f / (1.f + expf(-ig));
    const float sf = 1.f / (1.f + expf(-fg));
    const float so = 1.f / (1.f + expf(-og));
    const float cn = sf * c[b * H2_ + g] + si * tanhf(gg);
    const float hn = so * tanhf(cn);
    c[b * H2_ + g] = cn;
    h[b * H2_ + g] = hn;
    hbf[b * H2_ + g] = f32_to_bf16(hn);
  }
}

// logits[b, t, v] = h_new@W_out^T + b_out, written straight into d_out
__global__ void k_logits_gemm(const __bf16* __restrict__ Hb, const __bf16* __restrict__ Wout,
                              const float* __restrict__ b_out, float* __restrict__ out_logits, int t) {
  const int lane = threadIdx.x & 31;
  const int n0 = blockIdx.x * 64;
  const int m0 = blockIdx.y * 16;
  v8f acc[4] = {};
  gemm_kloop<4>(Hb, H2_, m0, Wout, H2_, n0, H2_, lane, acc);
  const int rb = m0 + ((lane >> 4) * 8);
#pragma unroll
  for (int j = 0; j < 4; ++j) {
    const int col = n0 + 16 * j + (lane & 15);
    const float bias = b_out[col];
#pragma unroll
    for (int r = 0; r < 8; ++r) {
      const int brow = rb + r;
      out_logits[((size_t)brow * T_ + t) * V_ + col] = acc[j][r] + bias;
    }
  }
}

__global__ void k_argmax(const float* __restrict__ out_logits, float* __restrict__ out_wid, int t) {
  const int b = blockIdx.x;
  const int tid = threadIdx.x;
  __shared__ float sv[256];
  __shared__ int   si[256];
  const float* row = out_logits + ((size_t)b * T_ + t) * V_;
  float best = -3.4e38f;
  int bi = 0;
  for (int v = tid; v < V_; v += 256) {
    const float x = row[v];
    if (x > best) { best = x; bi = v; }     // strictly >: keeps first max in stride
  }
  sv[tid] = best; si[tid] = bi;
  __syncthreads();
  for (int off = 128; off > 0; off >>= 1) {
    if (tid < off) {
      const float xv = sv[tid + off];
      const int   xi = si[tid + off];
      if (xv > sv[tid] || (xv == sv[tid] && xi < si[tid])) { sv[tid] = xv; si[tid] = xi; }
    }
    __syncthreads();
  }
  if (tid == 0) out_wid[b * T_ + t] = (float)si[0];
}

// ---------------- host launcher ----------------

extern "C" void kernel_launch(void* const* d_in, const int* in_sizes, int n_in,
                              void* d_out, int out_size, void* d_ws, size_t ws_size,
                              hipStream_t stream) {
  (void)in_sizes; (void)n_in; (void)out_size; (void)ws_size;

  const float* all_hidden = (const float*)d_in[0];   // [B,S,2H]
  const float* h0         = (const float*)d_in[1];   // [B,2H]
  const float* c0         = (const float*)d_in[2];   // [B,2H]
  const float* embedding  = (const float*)d_in[3];   // [V,E]
  const float* W_att      = (const float*)d_in[4];   // [2H,2H]
  const float* b_att      = (const float*)d_in[5];   // [2H]
  const float* W_ih       = (const float*)d_in[6];   // [8H, 2H+E]
  const float* b_ih       = (const float*)d_in[7];   // [8H]
  const float* W_hh       = (const float*)d_in[8];   // [8H, 2H]
  const float* b_hh       = (const float*)d_in[9];   // [8H]
  const float* W_out      = (const float*)d_in[10];  // [V, 2H]
  const float* b_out      = (const float*)d_in[11];  // [V]
  const int*   target     = (const int*)d_in[12];    // [B,T]

  float* out_logits = (float*)d_out;                        // [B,T,V]
  float* out_wid    = out_logits + (size_t)B_ * T_ * V_;    // [B,T]

  // workspace carve-up (256B aligned)
  char* ws = (char*)d_ws;
  size_t off = 0;
  auto carve = [&](size_t bytes) -> char* {
    char* p = ws + off;
    off = (off + bytes + 255) & ~(size_t)255;
    return p;
  };
  unsigned short* wout_bf = (unsigned short*)carve((size_t)V_ * H2_ * 2);   // 65.5 MB
  unsigned short* wih_bf  = (unsigned short*)carve((size_t)G4_ * X_ * 2);   // 12.6 MB
  unsigned short* whh_bf  = (unsigned short*)carve((size_t)G4_ * H2_ * 2);  //  8.4 MB
  unsigned short* watt_bf = (unsigned short*)carve((size_t)H2_ * H2_ * 2);  //  2.1 MB
  unsigned short* ah_bf   = (unsigned short*)carve((size_t)B_ * S_ * H2_ * 2); // 4.2 MB
  float*          attp    = (float*)carve((size_t)B_ * S_ * H2_ * 4);       //  8.4 MB
  float*          h_st    = (float*)carve((size_t)B_ * H2_ * 4);
  float*          c_st    = (float*)carve((size_t)B_ * H2_ * 4);
  unsigned short* h_bf    = (unsigned short*)carve((size_t)B_ * H2_ * 2);
  unsigned short* x_bf    = (unsigned short*)carve((size_t)B_ * X_ * 2);
  float*          gates   = (float*)carve((size_t)B_ * G4_ * 4);

  // one-time: weight / activation bf16 conversion (kept L2-resident thereafter)
  k_cvt_bf16<<<2048, 256, 0, stream>>>(W_out,      wout_bf, V_ * H2_);
  k_cvt_bf16<<<2048, 256, 0, stream>>>(W_ih,       wih_bf,  G4_ * X_);
  k_cvt_bf16<<<2048, 256, 0, stream>>>(W_hh,       whh_bf,  G4_ * H2_);
  k_cvt_bf16<<<1024, 256, 0, stream>>>(W_att,      watt_bf, H2_ * H2_);
  k_cvt_bf16<<<1024, 256, 0, stream>>>(all_hidden, ah_bf,   B_ * S_ * H2_);

  // hoisted attention projection: [2048,1024] x [1024,1024]^T
  k_attproj_gemm<<<dim3(H2_ / 64, (B_ * S_) / 16), 32, 0, stream>>>(
      (const __bf16*)ah_bf, (const __bf16*)watt_bf, b_att, attp);

  k_init_state<<<(B_ * H2_ + 255) / 256, 256, 0, stream>>>(h0, c0, h_st, c_st, h_bf);

  for (int t = 0; t < T_; ++t) {
    k_attention<<<B_, 256, 0, stream>>>(attp, all_hidden, h_st, embedding, target, x_bf, t);
    k_gates_gemm<<<dim3(G4_ / 64, B_ / 16), 32, 0, stream>>>(
        (const __bf16*)x_bf, (const __bf16*)h_bf,
        (const __bf16*)wih_bf, (const __bf16*)whh_bf, b_ih, b_hh, gates);
    k_lstm<<<B_, 256, 0, stream>>>(gates, h_st, c_st, h_bf);
    k_logits_gemm<<<dim3(V_ / 64, B_ / 16), 32, 0, stream>>>(
        (const __bf16*)h_bf, (const __bf16*)wout_bf, b_out, out_logits, t);
    k_argmax<<<B_, 256, 0, stream>>>(out_logits, out_wid, t);
  }
}
